// MultiscaleDeformableSelfAttention_6476810682802
// MI455X (gfx1250) — matrix operand order repaired and checked by reading, not compiled, not run on hardware
//
#include <hip/hip_runtime.h>

// ---------------------------------------------------------------------------
// Multi-scale deformable attention for gfx1250 (MI455X), wave32 + WMMA.
// B=4, N=8500, D=256, M=8 heads, D_HEAD=32, L=4 levels, K=4 points.
// ---------------------------------------------------------------------------

typedef _Float16 half_t;
typedef _Float16 v16h __attribute__((ext_vector_type(16)));
typedef _Float16 v8h  __attribute__((ext_vector_type(8)));
typedef float    v8f  __attribute__((ext_vector_type(8)));

union V16 { v16h v; v8h h[2]; };

#define B_SZ     4
#define N_PTS    8500
#define BN_ROWS  (B_SZ * N_PTS)     // 34000
#define D_MODEL  256
#define N_HEADS  8
#define D_HEAD   32
#define N_LVL    4
#define N_PTK    4

// ---------------------------------------------------------------------------
// h = hidden + pos, cast to f16
// ---------------------------------------------------------------------------
__global__ void addcast_kernel(const float* __restrict__ a,
                               const float* __restrict__ b,
                               half_t* __restrict__ o, int total) {
    int i = blockIdx.x * blockDim.x + threadIdx.x;
    if (i < total) o[i] = (half_t)(a[i] + b[i]);
}

// ---------------------------------------------------------------------------
// Weight transpose + cast: W[k=256, nout] (row-major) -> WT[n, k=256] f16
// so that B fragments (16 consecutive K of one column) are contiguous 32B.
// ---------------------------------------------------------------------------
__global__ void wcast_kernel(const float* __restrict__ W,
                             half_t* __restrict__ WT, int nout) {
    int tid = blockIdx.x * blockDim.x + threadIdx.x;
    if (tid >= 256 * nout) return;
    int n = tid >> 8;       // output column
    int k = tid & 255;      // reduction index
    WT[tid] = (half_t)W[k * nout + n];
}

// ---------------------------------------------------------------------------
// WMMA GEMM: C[rows=strips*16, nout] = A_f16[rows,256] @ WT^T + bias
// One wave per 16-row strip x nt column-tiles. A fragments (full K=256)
// kept in registers across the N-tile loop.
// ---------------------------------------------------------------------------
template<bool OUT_HALF>
__global__ void gemm_wmma_kernel(const half_t* __restrict__ A,
                                 const half_t* __restrict__ WT,
                                 const float*  __restrict__ bias,
                                 float*  __restrict__ Cf,
                                 half_t* __restrict__ Ch,
                                 int strips, int nout, int ntgroups, int nt)
{
    int gid  = blockIdx.x * blockDim.x + threadIdx.x;
    int wave = gid >> 5;
    int lane = gid & 31;
    if (wave >= strips * ntgroups) return;   // wave-uniform exit

    int strip   = wave / ntgroups;
    int ng      = wave - strip * ntgroups;
    int rowbase = strip << 4;
    int lrow    = lane & 15;
    int khalf   = lane >> 4;                 // 0: K {0..15}, 1: K {16..31}

    // Load all A fragments for K = 0..255 (8 ktiles of 32)
    V16 a[8];
    const half_t* arow = A + (size_t)(rowbase + lrow) * D_MODEL + (khalf << 4);
    #pragma unroll
    for (int kt = 0; kt < 8; ++kt) {
        a[kt].v = *(const v16h*)(arow + kt * 32);
    }

    for (int t = 0; t < nt; ++t) {
        int nbase = ((ng * nt) + t) << 4;
        int ncol  = nbase + lrow;            // this lane's output column
        const half_t* wrow = WT + (size_t)ncol * D_MODEL + (khalf << 4);

        v8f c = {};
        #pragma unroll
        for (int kt = 0; kt < 8; ++kt) {
            v16h b = *(const v16h*)(wrow + kt * 32);
            c = __builtin_amdgcn_wmma_f32_16x16x32_f16(
                    false, a[kt].v, false, b, (short)0, c, false, false);
        }

        float bv    = bias[ncol];
        int   rbase = rowbase + (khalf << 3);   // c[i] holds row rbase+i
        #pragma unroll
        for (int i = 0; i < 8; ++i) {
            size_t off = (size_t)(rbase + i) * nout + ncol;
            float  val = c[i] + bv;
            if constexpr (OUT_HALF) Ch[off] = (half_t)val;
            else                    Cf[off] = val;
        }
    }
}

// ---------------------------------------------------------------------------
// Sampling + softmax + weighted sum. One wave per (b,n,m); lane = head dim.
// Gathers are 64B-coalesced (32 lanes x f16 over D_HEAD).
// ---------------------------------------------------------------------------
__global__ void msda_sample_kernel(const half_t* __restrict__ vhalf,  // [BN,256] f16
                                   const float*  __restrict__ offb,   // [BN,256]
                                   const float*  __restrict__ attn,   // [BN,128]
                                   const float*  __restrict__ ref,    // [BN,L,2]
                                   half_t* __restrict__ mid)          // [BN,256] f16
{
    int gid  = blockIdx.x * blockDim.x + threadIdx.x;
    int wave = gid >> 5;
    int d    = gid & 31;                  // head dim
    int m    = wave & (N_HEADS - 1);
    int bn   = wave >> 3;                 // b*N + n
    if (bn >= BN_ROWS) return;
    int b = bn / N_PTS;

    // softmax over the 16 logits of this head (computed redundantly per lane)
    const float* lg = attn + (size_t)bn * 128 + m * 16;
    float ew[16];
    float mx = lg[0];
    #pragma unroll
    for (int j = 1; j < 16; ++j) mx = fmaxf(mx, lg[j]);
    float ssum = 0.f;
    #pragma unroll
    for (int j = 0; j < 16; ++j) { ew[j] = expf(lg[j] - mx); ssum += ew[j]; }
    float inv = 1.0f / ssum;

    const float* ofr = offb + (size_t)bn * 256 + m * 32;  // [(l*4+k)*2 + c]
    const float* rfr = ref  + (size_t)bn * 8;             // [l*2 + c]

    const int Hs[N_LVL] = {80, 40, 20, 10};
    const int St[N_LVL] = {0, 6400, 8000, 8400};

    float acc = 0.f;
    #pragma unroll
    for (int l = 0; l < N_LVL; ++l) {
        const int   Hl = Hs[l], Wl = Hs[l];          // square levels
        const float Hf = (float)Hl, Wf = (float)Wl;
        const int   base_row = b * N_PTS + St[l];
        const float ry = rfr[l * 2 + 0];             // comp0 pairs with H
        const float rx = rfr[l * 2 + 1];             // comp1 pairs with W
        #pragma unroll
        for (int k = 0; k < N_PTK; ++k) {
            const int j  = l * 4 + k;
            const float oy = ofr[j * 2 + 0];
            const float ox = ofr[j * 2 + 1];
            // sp = ref + off/shape, then (..., ::-1): x <- comp1, y <- comp0
            const float gx = rx + ox / Wf;
            const float gy = ry + oy / Hf;
            // grid_sample, align_corners=False
            const float x = (gx + 1.0f) * (0.5f * Wf) - 0.5f;
            const float y = (gy + 1.0f) * (0.5f * Hf) - 0.5f;
            const float x0f = floorf(x), y0f = floorf(y);
            const float wx = x - x0f, wy = y - y0f;
            const int x0 = (int)x0f, y0 = (int)y0f;

            auto tap = [&](int xi, int yi) -> float {
                bool valid = (xi >= 0) & (xi < Wl) & (yi >= 0) & (yi < Hl);
                int xc = xi < 0 ? 0 : (xi > Wl - 1 ? Wl - 1 : xi);
                int yc = yi < 0 ? 0 : (yi > Hl - 1 ? Hl - 1 : yi);
                float v = (float)vhalf[(size_t)(base_row + yc * Wl + xc) * D_MODEL
                                       + m * D_HEAD + d];
                return valid ? v : 0.0f;
            };
            const float v00 = tap(x0,     y0);
            const float v10 = tap(x0 + 1, y0);
            const float v01 = tap(x0,     y0 + 1);
            const float v11 = tap(x0 + 1, y0 + 1);
            const float sv = v00 * (1.f - wx) * (1.f - wy)
                           + v10 * wx * (1.f - wy)
                           + v01 * (1.f - wx) * wy
                           + v11 * wx * wy;
            acc += ew[j] * sv;
        }
    }
    mid[(size_t)bn * 256 + m * D_HEAD + d] = (half_t)(acc * inv);
}

// ---------------------------------------------------------------------------
extern "C" void kernel_launch(void* const* d_in, const int* in_sizes, int n_in,
                              void* d_out, int out_size, void* d_ws, size_t ws_size,
                              hipStream_t stream) {
    const float* hidden = (const float*)d_in[0];
    const float* pos    = (const float*)d_in[1];
    const float* ref    = (const float*)d_in[2];
    // d_in[3] spatial_shapes: compile-time constants, unused
    const float* Wv   = (const float*)d_in[4];
    const float* bv   = (const float*)d_in[5];
    const float* Woff = (const float*)d_in[6];
    const float* boff = (const float*)d_in[7];
    const float* Wa   = (const float*)d_in[8];
    const float* ba   = (const float*)d_in[9];
    const float* Wout = (const float*)d_in[10];
    const float* bout = (const float*)d_in[11];
    float* out = (float*)d_out;

    char*  ws  = (char*)d_ws;
    size_t cur = 0;
    auto alloc = [&](size_t bytes) {
        void* p = ws + cur;
        cur = (cur + bytes + 255) & ~(size_t)255;
        return p;
    };
    half_t* h_half   = (half_t*)alloc((size_t)BN_ROWS * 256 * sizeof(half_t));
    half_t* v_half   = (half_t*)alloc((size_t)BN_ROWS * 256 * sizeof(half_t));
    half_t* mid_half = (half_t*)alloc((size_t)BN_ROWS * 256 * sizeof(half_t));
    float*  off_f    = (float*) alloc((size_t)BN_ROWS * 256 * sizeof(float));
    float*  attn_f   = (float*) alloc((size_t)BN_ROWS * 128 * sizeof(float));
    half_t* wvT      = (half_t*)alloc(256 * 256 * sizeof(half_t));
    half_t* woffT    = (half_t*)alloc(256 * 256 * sizeof(half_t));
    half_t* waT      = (half_t*)alloc(128 * 256 * sizeof(half_t));
    half_t* woutT    = (half_t*)alloc(256 * 256 * sizeof(half_t));

    const int strips = BN_ROWS / 16;   // 2125

    // 1) h = hidden + pos (f16)
    {
        int total = BN_ROWS * 256;
        addcast_kernel<<<(total + 255) / 256, 256, 0, stream>>>(hidden, pos, h_half, total);
    }
    // 2) weight transpose + cast
    wcast_kernel<<<(256 * 256 + 255) / 256, 256, 0, stream>>>(Wv,   wvT,   256);
    wcast_kernel<<<(256 * 256 + 255) / 256, 256, 0, stream>>>(Woff, woffT, 256);
    wcast_kernel<<<(256 * 128 + 255) / 256, 256, 0, stream>>>(Wa,   waT,   128);
    wcast_kernel<<<(256 * 256 + 255) / 256, 256, 0, stream>>>(Wout, woutT, 256);

    // 3) values = h @ Wv + bv  (f16 output, nout=256: 4 groups x 4 tiles)
    {
        int waves = strips * 4, threads = waves * 32;
        gemm_wmma_kernel<true><<<(threads + 255) / 256, 256, 0, stream>>>(
            h_half, wvT, bv, nullptr, v_half, strips, 256, 4, 4);
    }
    // 4) off = h @ Woff + boff (f32)
    {
        int waves = strips * 4, threads = waves * 32;
        gemm_wmma_kernel<false><<<(threads + 255) / 256, 256, 0, stream>>>(
            h_half, woffT, boff, off_f, nullptr, strips, 256, 4, 4);
    }
    // 5) attn logits = h @ Wa + ba (f32, nout=128: 2 groups x 4 tiles)
    {
        int waves = strips * 2, threads = waves * 32;
        gemm_wmma_kernel<false><<<(threads + 255) / 256, 256, 0, stream>>>(
            h_half, waT, ba, attn_f, nullptr, strips, 128, 2, 4);
    }
    // 6) sampling + softmax + weighted sum -> mid (f16)
    {
        int waves = BN_ROWS * N_HEADS, threads = waves * 32;  // 8.7M threads
        msda_sample_kernel<<<(threads + 255) / 256, 256, 0, stream>>>(
            v_half, off_f, attn_f, ref, mid_half);
    }
    // 7) out = mid @ Wout + bout (f32 -> d_out)
    {
        int waves = strips * 4, threads = waves * 32;
        gemm_wmma_kernel<false><<<(threads + 255) / 256, 256, 0, stream>>>(
            mid_half, woutT, bout, out, nullptr, strips, 256, 4, 4);
    }
}
